// TCA_309237645935
// MI455X (gfx1250) — matrix operand order repaired and checked by reading, not compiled, not run on hardware
//
#include <hip/hip_runtime.h>
#include <hip/hip_bf16.h>

// ---------- types ----------
typedef __attribute__((ext_vector_type(16))) __bf16         v16bf;
typedef __attribute__((ext_vector_type(8)))  float          v8f;
typedef __attribute__((ext_vector_type(8)))  unsigned short ushort8;
typedef __attribute__((ext_vector_type(4)))  unsigned int   uint4v;
typedef __attribute__((ext_vector_type(8)))  int            int8v;
typedef __attribute__((ext_vector_type(4)))  int            int4v;

union FragB16 {
    v16bf          bf;
    ushort8        h[2];
    unsigned short u[16];
};

__device__ __forceinline__ unsigned short f2bf(float f) {
    unsigned int x = __float_as_uint(f);
    unsigned int r = x + 0x7fffu + ((x >> 16) & 1u);   // RNE
    return (unsigned short)(r >> 16);
}

__device__ __forceinline__ v8f wmma_bf16(const FragB16& a, const FragB16& b, v8f c) {
    // D = A(16x32 bf16) * B(32x16 bf16) + C(16x16 f32)
    return __builtin_amdgcn_wmma_f32_16x16x32_bf16(
        false, a.bf, false, b.bf, (short)0, c, false, false);
}

// ---------- TDM availability (device pass only; manual fallback otherwise) ----
#if defined(__HIP_DEVICE_COMPILE__) && defined(__has_builtin)
#if __has_builtin(__builtin_amdgcn_tensor_load_to_lds) && __has_builtin(__builtin_amdgcn_s_wait_tensorcnt)
#define USE_TDM 1
#endif
#endif
#ifndef USE_TDM
#define USE_TDM 0
#endif

#if USE_TDM
// 2D TDM load: tile (tile_d1 rows x tile_d0 elems) of 2-byte data from global
// to LDS, with optional LDS row padding. D# packing per CDNA5 ISA 08 §8.3/8.4.
// This toolchain exposes the 6-arg builtin:
//   (uint32x4 g0, int32x8 g1, int32x4 g2, int32x4 g3, int32x8 gx, i32 cpol)
__device__ __forceinline__ void tdm_load_2d_b16(
        unsigned lds_addr, const void* gptr,
        unsigned tensor_d0, unsigned tensor_d1, unsigned long long stride0,
        unsigned tile_d0, unsigned tile_d1,
        unsigned pad_interval_code, unsigned pad_amount_code, unsigned pad_en)
{
    unsigned long long ga = (unsigned long long)(size_t)gptr;
    uint4v g0;
    g0[0] = 1u;                                             // count=1, user mode
    g0[1] = lds_addr;                                       // lds_addr [63:32]
    g0[2] = (unsigned)(ga & 0xFFFFFFFFu);                   // global_addr lo
    g0[3] = (unsigned)((ga >> 32) & 0x01FFFFFFu) | (2u << 30);  // addr hi | type=2

    int8v g1;
    unsigned w0 = (1u << 16)                                // data_size=1 (2B)
                | (pad_en << 20)
                | (pad_interval_code << 22)
                | (pad_amount_code << 25);
    g1[0] = (int)w0;
    g1[1] = (int)((tensor_d0 & 0xFFFFu) << 16);             // dim0[15:0] @ [63:48]
    g1[2] = (int)(((tensor_d0 >> 16) & 0xFFFFu) | ((tensor_d1 & 0xFFFFu) << 16));
    g1[3] = (int)(((tensor_d1 >> 16) & 0xFFFFu) | ((tile_d0 & 0xFFFFu) << 16));
    g1[4] = (int)(tile_d1 & 0xFFFFu);                       // tile_dim1 | tile_dim2=0
    g1[5] = (int)(unsigned)(stride0 & 0xFFFFFFFFu);         // dim0_stride lo
    g1[6] = (int)(unsigned)((stride0 >> 32) & 0xFFFFu);     // stride hi | dim1_stride lo=0
    g1[7] = 0;

    int4v gz4 = {0, 0, 0, 0};
    int8v gz8 = {0, 0, 0, 0, 0, 0, 0, 0};
    __builtin_amdgcn_tensor_load_to_lds(g0, g1, gz4, gz4, gz8, 0);
}
#endif

// ---------- f32 -> bf16 conversion ----------
__global__ __launch_bounds__(256)
void f32_to_bf16_kernel(const float* __restrict__ src, unsigned short* __restrict__ dst, int n) {
    int i = blockIdx.x * 256 + threadIdx.x;
    if (i < n) dst[i] = f2bf(src[i]);
}

// ---------- bf16 GEMM: C[M,N] = A[M,K] @ W[K,N] + bias ----------
// 256 thr = 8 waves; block tile 128x128; wave tile 16x128; k-step 32.
template<bool BF16OUT>
__global__ __launch_bounds__(256)
void gemm_bf16_kernel(const unsigned short* __restrict__ A,
                      const unsigned short* __restrict__ W,
                      const float* __restrict__ bias,
                      unsigned short* __restrict__ outBf,
                      float* __restrict__ outF,
                      int M, int N, int K)
{
    __shared__ __align__(16) unsigned short lWT[128][40];  // W tile transposed [n][k]
    const int tid  = threadIdx.x;
    const int w    = tid >> 5;
    const int lane = tid & 31;
    const int lm   = lane & 15;
    const int hi   = lane >> 4;
    const int m0   = blockIdx.x * 128 + w * 16;
    const int n0   = blockIdx.y * 128;

    const v8f vzero = {0.f,0.f,0.f,0.f,0.f,0.f,0.f,0.f};
    v8f acc[8];
#pragma unroll
    for (int j = 0; j < 8; ++j) acc[j] = vzero;

    for (int kc = 0; kc < K; kc += 32) {
        __syncthreads();
        // stage 32x128 W tile transposed (coalesced global read)
#pragma unroll
        for (int i = 0; i < 16; ++i) {
            int e  = tid + 256 * i;        // 4096 elements
            int kk = e >> 7;
            int nn = e & 127;
            lWT[nn][kk] = W[(size_t)(kc + kk) * N + n0 + nn];
        }
        // prefetch next W tile into L2 (global_prefetch_b8)
        if (kc + 32 < K)
            __builtin_prefetch(W + (size_t)(kc + 32 + (tid >> 7)) * N + n0 + (tid & 127), 0, 1);
        __syncthreads();

        // A fragment (16x32): elem e -> k = hi*16 + e
        FragB16 fa;
        const unsigned short* arow = A + (size_t)(m0 + lm) * K + kc + hi * 16;
        fa.h[0] = *(const ushort8*)(arow);
        fa.h[1] = *(const ushort8*)(arow + 8);

        // ping-pong B fragments so ds_loads for j+1 overlap wmma j
        FragB16 fb[2];
        {
            const unsigned short* bp = &lWT[lm][hi * 16];
            fb[0].h[0] = *(const ushort8*)(bp);
            fb[0].h[1] = *(const ushort8*)(bp + 8);
        }
#pragma unroll
        for (int j = 0; j < 8; ++j) {
            if (j < 7) {
                const unsigned short* bp = &lWT[(j + 1) * 16 + lm][hi * 16];
                fb[(j + 1) & 1].h[0] = *(const ushort8*)(bp);
                fb[(j + 1) & 1].h[1] = *(const ushort8*)(bp + 8);
            }
            acc[j] = wmma_bf16(fa, fb[j & 1], acc[j]);
        }
    }

    // epilogue: C layout row = r + 8*hi, col = lane%16 — no runtime branches
#pragma unroll
    for (int j = 0; j < 8; ++j) {
        int n = n0 + j * 16 + lm;
        float bv = bias[n];
#pragma unroll
        for (int r = 0; r < 8; ++r) {
            int m = m0 + r + 8 * hi;
            float v = acc[j][r] + bv;
            if (BF16OUT) outBf[(size_t)m * N + n] = f2bf(v);
            else         outF [(size_t)m * N + n] = v;
        }
    }
}

// ---------- fused dual-softmax flash attention ----------
// grid = (64 pairs, 8 query tiles); 256 thr = 8 waves; wave: 16 queries x dv=64.
__global__ __launch_bounds__(256)
void attn_dual_kernel(const unsigned short* __restrict__ Q,
                      const unsigned short* __restrict__ Km,
                      const unsigned short* __restrict__ Vm,
                      const float* __restrict__ adj,
                      const int*   __restrict__ mask,
                      const float* __restrict__ alphaPtr,
                      float* __restrict__ Xo)
{
    // V tile row-major [key][dim] with 8-ushort row pad (72 pitch = TDM pad 4 DW / 32 DW)
    __shared__ __align__(16) unsigned short lV[32][72];
    __shared__ __align__(16) unsigned short lPG[8][16][40];
    __shared__ __align__(16) unsigned short lPL[8][16][40];

    const int tid  = threadIdx.x;
    const int w    = tid >> 5;
    const int lane = tid & 31;
    const int lm   = lane & 15;
    const int hi   = lane >> 4;
    const int p    = blockIdx.x;                 // head*4 + batch
    const int q0   = blockIdx.y * 128 + w * 16;  // query row within pair

    const unsigned short* Qp = Q  + (size_t)p * 1024 * 64;
    const unsigned short* Kp = Km + (size_t)p * 1024 * 64;
    const unsigned short* Vp = Vm + (size_t)p * 1024 * 64;

    // Q strip 16x64 as two A fragments
    FragB16 fq0, fq1;
    {
        const unsigned short* qrow = Qp + (size_t)(q0 + lm) * 64;
        fq0.h[0] = *(const ushort8*)(qrow +      hi * 16);
        fq0.h[1] = *(const ushort8*)(qrow +      hi * 16 + 8);
        fq1.h[0] = *(const ushort8*)(qrow + 32 + hi * 16);
        fq1.h[1] = *(const ushort8*)(qrow + 32 + hi * 16 + 8);
    }

    const v8f vzero = {0.f,0.f,0.f,0.f,0.f,0.f,0.f,0.f};
    float mG[8], mL[8], sG[8], sL[8];
    v8f oG[4], oL[4];
#pragma unroll
    for (int r = 0; r < 8; ++r) { mG[r] = -3.0e38f; mL[r] = -3.0e38f; sG[r] = 0.f; sL[r] = 0.f; }
#pragma unroll
    for (int j = 0; j < 4; ++j) { oG[j] = vzero; oL[j] = vzero; }

    const float norm = 0.03125f;  // 1/sqrt(1024)

    for (int kt = 0; kt < 32; ++kt) {
        const int kb = kt * 32;

        __syncthreads();   // prior iteration's lV reads complete
#if USE_TDM
        if (w == 0) {
            // async DMA of 32x64 bf16 V tile into LDS, padded rows (pitch 72 ushort)
            tdm_load_2d_b16((unsigned)(size_t)&lV[0][0], Vp + (size_t)kb * 64,
                            /*tensor_d0=*/64, /*tensor_d1=*/32, /*stride0=*/64,
                            /*tile_d0=*/64, /*tile_d1=*/32,
                            /*pad_interval(32DW)=*/4u, /*pad_amount(4DW)=*/3u, 1u);
            __builtin_amdgcn_s_wait_tensorcnt((short)0);
        }
#else
#pragma unroll
        for (int i = 0; i < 8; ++i) {
            int e   = tid + 256 * i;   // 2048 elements
            int key = e >> 6;
            int d   = e & 63;
            lV[key][d] = Vp[(size_t)(kb + key) * 64 + d];
        }
#endif
        // prefetch next tile's K rows / adj / mask into L2
        if (kt + 1 < 32) {
            __builtin_prefetch(Kp + (size_t)(kb + 32 + lane) * 64, 0, 1);
            __builtin_prefetch(adj  + (size_t)(q0 + lm) * 1024 + kb + 32, 0, 1);
            __builtin_prefetch(mask + (size_t)(q0 + lm) * 1024 + kb + 32, 0, 1);
        }
        __syncthreads();

        // ----- scores: two 16-key tiles, contraction over 64 features -----
        v8f sc[2] = { vzero, vzero };
#pragma unroll
        for (int g = 0; g < 2; ++g) {
            const unsigned short* kr = Kp + (size_t)(kb + g * 16 + lm) * 64;
            FragB16 fk0, fk1;   // B: elem e -> feat = c*32 + 16*hi + e, key = lm
            fk0.h[0] = *(const ushort8*)(kr +      hi * 16);
            fk0.h[1] = *(const ushort8*)(kr +      hi * 16 + 8);
            fk1.h[0] = *(const ushort8*)(kr + 32 + hi * 16);
            fk1.h[1] = *(const ushort8*)(kr + 32 + hi * 16 + 8);
            sc[g] = wmma_bf16(fq0, fk0, sc[g]);
            sc[g] = wmma_bf16(fq1, fk1, sc[g]);
        }

        // ----- logits (f32) -----
        float gv[2][8], lv[2][8];
#pragma unroll
        for (int g = 0; g < 2; ++g) {
            int kcol = kb + g * 16 + lm;
#pragma unroll
            for (int r = 0; r < 8; ++r) {
                int qr = q0 + r + 8 * hi;
                float av = adj [(size_t)qr * 1024 + kcol];
                int   mk = mask[(size_t)qr * 1024 + kcol];
                float gvv = sc[g][r] * norm + av;
                gv[g][r] = gvv;
                lv[g][r] = (mk == 0) ? -1.0e9f : gvv;
            }
        }

        // ----- online dual softmax (reductions within 16-lane halves) -----
#pragma unroll
        for (int r = 0; r < 8; ++r) {
            float gm  = fmaxf(gv[0][r], gv[1][r]);
            float lmx = fmaxf(lv[0][r], lv[1][r]);
#pragma unroll
            for (int off = 1; off < 16; off <<= 1) {
                gm  = fmaxf(gm,  __shfl_xor(gm,  off, 16));
                lmx = fmaxf(lmx, __shfl_xor(lmx, off, 16));
            }
            float nmG = fmaxf(mG[r], gm);
            float nmL = fmaxf(mL[r], lmx);
            float scG = __expf(mG[r] - nmG);
            float scL = __expf(mL[r] - nmL);
            mG[r] = nmG; mL[r] = nmL;
            sG[r] *= scG; sL[r] *= scL;
#pragma unroll
            for (int j = 0; j < 4; ++j) { oG[j][r] *= scG; oL[j][r] *= scL; }

            float pg0 = __expf(gv[0][r] - nmG), pg1 = __expf(gv[1][r] - nmG);
            float pl0 = __expf(lv[0][r] - nmL), pl1 = __expf(lv[1][r] - nmL);
            float ssg = pg0 + pg1, ssl = pl0 + pl1;
#pragma unroll
            for (int off = 1; off < 16; off <<= 1) {
                ssg += __shfl_xor(ssg, off, 16);
                ssl += __shfl_xor(ssl, off, 16);
            }
            sG[r] += ssg; sL[r] += ssl;

            int row = r + 8 * hi;
            lPG[w][row][lm]      = f2bf(pg0);
            lPG[w][row][16 + lm] = f2bf(pg1);
            lPL[w][row][lm]      = f2bf(pl0);
            lPL[w][row][16 + lm] = f2bf(pl1);
        }

        // ----- P (C-layout -> A-layout via LDS) and AV WMMAs -----
        FragB16 fpg, fpl;
        fpg.h[0] = *(const ushort8*)&lPG[w][lm][hi * 16];
        fpg.h[1] = *(const ushort8*)&lPG[w][lm][hi * 16 + 8];
        fpl.h[0] = *(const ushort8*)&lPL[w][lm][hi * 16];
        fpl.h[1] = *(const ushort8*)&lPL[w][lm][hi * 16 + 8];

        FragB16 fv[2];
        // B frag from row-major V tile: elem e -> key = 16*hi+e, dim = j*16+lm
#pragma unroll
        for (int e = 0; e < 16; ++e) fv[0].u[e] = lV[16 * hi + e][lm];
#pragma unroll
        for (int j = 0; j < 4; ++j) {
            if (j < 3) {
#pragma unroll
                for (int e = 0; e < 16; ++e)
                    fv[(j + 1) & 1].u[e] = lV[16 * hi + e][(j + 1) * 16 + lm];
            }
            oG[j] = wmma_bf16(fpg, fv[j & 1], oG[j]);
            oL[j] = wmma_bf16(fpl, fv[j & 1], oL[j]);
        }
    }

    // ----- combine and store f32 -----
    float al = alphaPtr[0];
    float aa = 1.f / (1.f + __expf(-al));
#pragma unroll
    for (int j = 0; j < 4; ++j) {
#pragma unroll
        for (int r = 0; r < 8; ++r) {
            int qr = q0 + r + 8 * hi;
            float og = oG[j][r] / sG[r];
            float ol = oL[j][r] / sL[r];
            Xo[((size_t)p * 1024 + qr) * 64 + j * 16 + lm] = aa * og + (1.f - aa) * ol;
        }
    }
}

// ---------- signed sqrt + L2 normalize rows of [4096,1024], emit bf16 ----------
__global__ __launch_bounds__(256)
void ssqrt_norm_kernel(const float* __restrict__ Xo, unsigned short* __restrict__ Xob)
{
    __shared__ float red[8];
    const int row  = blockIdx.x;
    const int tid  = threadIdx.x;
    const int wid  = tid >> 5;
    const int lane = tid & 31;
    const float* x = Xo + (size_t)row * 1024;

    float y[4];
    float ssum = 0.f;
#pragma unroll
    for (int i = 0; i < 4; ++i) {
        float v  = x[tid + 256 * i];
        float yv = sqrtf(fmaxf(v, 0.f)) - sqrtf(fmaxf(-v, 0.f));
        y[i] = yv;
        ssum += yv * yv;
    }
#pragma unroll
    for (int off = 1; off < 32; off <<= 1) ssum += __shfl_xor(ssum, off, 32);
    if (lane == 0) red[wid] = ssum;
    __syncthreads();
    if (tid == 0) {
        float t = 0.f;
#pragma unroll
        for (int k = 0; k < 8; ++k) t += red[k];
        red[0] = t;
    }
    __syncthreads();
    float inv = 1.f / fmaxf(sqrtf(red[0]), 1e-12f);
#pragma unroll
    for (int i = 0; i < 4; ++i)
        Xob[(size_t)row * 1024 + tid + 256 * i] = f2bf(y[i] * inv);
}

// ---------- host launch ----------
extern "C" void kernel_launch(void* const* d_in, const int* in_sizes, int n_in,
                              void* d_out, int out_size, void* d_ws, size_t ws_size,
                              hipStream_t stream)
{
    const float* x     = (const float*)d_in[0];
    const int*   mask  = (const int*)  d_in[1];
    const float* adj   = (const float*)d_in[2];
    const float* Wq    = (const float*)d_in[3];
    const float* bq    = (const float*)d_in[4];
    const float* Wk    = (const float*)d_in[5];
    const float* bk    = (const float*)d_in[6];
    const float* Wv    = (const float*)d_in[7];
    const float* bv    = (const float*)d_in[8];
    const float* Wo    = (const float*)d_in[9];
    const float* bo    = (const float*)d_in[10];
    const float* alpha = (const float*)d_in[11];
    float* out = (float*)d_out;

    const size_t MT = 4096;          // B*T
    const size_t D  = 1024;

    char* ws = (char*)d_ws;
    unsigned short* xb  = (unsigned short*)ws; ws += MT * D * 2;        // 8 MB
    unsigned short* wqb = (unsigned short*)ws; ws += D * D * 2;         // 2 MB
    unsigned short* wkb = (unsigned short*)ws; ws += D * D * 2;
    unsigned short* wvb = (unsigned short*)ws; ws += D * D * 2;
    unsigned short* wob = (unsigned short*)ws; ws += D * D * 2;
    unsigned short* qb  = (unsigned short*)ws; ws += MT * D * 2;        // 8 MB
    unsigned short* kb  = (unsigned short*)ws; ws += MT * D * 2;
    unsigned short* vb  = (unsigned short*)ws; ws += MT * D * 2;
    float*          xo  = (float*)ws;          ws += MT * D * 4;        // 16 MB
    unsigned short* xob = (unsigned short*)ws; ws += MT * D * 2;        // 8 MB

    // 1) f32 -> bf16
    f32_to_bf16_kernel<<<(int)(MT * D / 256), 256, 0, stream>>>(x,  xb,  (int)(MT * D));
    f32_to_bf16_kernel<<<(int)(D * D / 256),  256, 0, stream>>>(Wq, wqb, (int)(D * D));
    f32_to_bf16_kernel<<<(int)(D * D / 256),  256, 0, stream>>>(Wk, wkb, (int)(D * D));
    f32_to_bf16_kernel<<<(int)(D * D / 256),  256, 0, stream>>>(Wv, wvb, (int)(D * D));
    f32_to_bf16_kernel<<<(int)(D * D / 256),  256, 0, stream>>>(Wo, wob, (int)(D * D));

    // 2) QKV projections (bf16 out)
    dim3 gg(32, 8);
    gemm_bf16_kernel<true><<<gg, 256, 0, stream>>>(xb, wqb, bq, qb, nullptr, 4096, 1024, 1024);
    gemm_bf16_kernel<true><<<gg, 256, 0, stream>>>(xb, wkb, bk, kb, nullptr, 4096, 1024, 1024);
    gemm_bf16_kernel<true><<<gg, 256, 0, stream>>>(xb, wvb, bv, vb, nullptr, 4096, 1024, 1024);

    // 3) fused dual-softmax attention
    dim3 ga(64, 8);
    attn_dual_kernel<<<ga, 256, 0, stream>>>(qb, kb, vb, adj, mask, alpha, xo);

    // 4) signed sqrt + L2 normalize
    ssqrt_norm_kernel<<<4096, 256, 0, stream>>>(xo, xob);

    // 5) output projection (f32 out)
    gemm_bf16_kernel<false><<<gg, 256, 0, stream>>>(xob, wob, bo, nullptr, out, 4096, 1024, 1024);
}
